// AFT_FULL_89455578841517
// MI455X (gfx1250) — matrix operand order repaired and checked by reading, not compiled, not run on hardware
//
#include <hip/hip_runtime.h>

typedef __attribute__((ext_vector_type(16))) _Float16 v16h;
typedef __attribute__((ext_vector_type(8)))  _Float16 v8h;
typedef __attribute__((ext_vector_type(2)))  _Float16 v2h;
typedef __attribute__((ext_vector_type(8)))  float    v8f;
typedef __attribute__((ext_vector_type(4)))  int      v4i;

constexpr int DM    = 512;    // d_model
constexpr int NSEQ  = 2048;   // sequence length
constexpr int NB    = 16;     // batch
constexpr int KSTEP = 32;     // WMMA K per step (f16 16x16x32)
constexpr int MT    = 128;    // block tile rows
constexpr int NT    = 64;     // block tile cols
constexpr int LDA   = KSTEP + 8;  // LDS row stride in halves (80B, 16B-aligned)

#define CAT16 0,1,2,3,4,5,6,7,8,9,10,11,12,13,14,15

// gfx1250 async global->LDS copy (ASYNCcnt path), guarded so the file always compiles
#if defined(__has_builtin)
#  if __has_builtin(__builtin_amdgcn_global_load_async_to_lds_b128) && \
      __has_builtin(__builtin_amdgcn_s_wait_asynccnt)
#    define HAVE_ASYNC_LDS 1
#  endif
#endif
#ifndef HAVE_ASYNC_LDS
#  define HAVE_ASYNC_LDS 0
#endif

#if HAVE_ASYNC_LDS
typedef __attribute__((address_space(1))) v4i gv4i;   // global
typedef __attribute__((address_space(3))) v4i lv4i;   // LDS
#endif

// ---------------------------------------------------------------------------
// Kernel 1: fused QKV projection GEMM. [32768,512] x [512,512], z selects W.
//   z=0: sigmoid(x@Wq+bq) -> d_out (staging, overwritten by aft epilogue)
//   z=1: x@Wk+bk -> kbuf (f16)     z=2: x@Wv+bv -> vbuf (f16)
// x re-reads across col-blocks stay L2-resident (64MB << 192MB L2).
// ---------------------------------------------------------------------------
__global__ void __launch_bounds__(256)
qkv_kernel(const float* __restrict__ x,
           const float* __restrict__ Wq, const float* __restrict__ bq,
           const float* __restrict__ Wk, const float* __restrict__ bk,
           const float* __restrict__ Wv, const float* __restrict__ bv,
           float* __restrict__ sigq,
           _Float16* __restrict__ kbuf, _Float16* __restrict__ vbuf)
{
    __shared__ alignas(16) _Float16 As[MT][LDA];   // x tile, row-major
    __shared__ alignas(16) _Float16 Bs[NT][LDA];   // W tile, transposed (n-major)

    const int tid  = threadIdx.x;
    const int lane = tid & 31;
    const int wave = tid >> 5;
    const int wm   = wave & 3;    // 4 m-subtiles of 32 rows
    const int wn   = wave >> 2;   // 2 n-subtiles of 32 cols

    const int row0 = blockIdx.x * MT;
    const int col0 = blockIdx.y * NT;
    const int mode = blockIdx.z;
    const float* W    = (mode == 0) ? Wq : (mode == 1) ? Wk : Wv;
    const float* bias = (mode == 0) ? bq : (mode == 1) ? bk : bv;

    v8f acc[2][2] = {};

    const int kq = lane >> 4;     // K half-group select
    const int lr = lane & 15;

    for (int k0 = 0; k0 < DM; k0 += KSTEP) {
        // stage A: x[row0..+128][k0..+32] (f32 -> f16)
        #pragma unroll
        for (int r = 0; r < 16; ++r) {
            int idx = tid + r * 256;            // 0..4095
            int i   = idx >> 5;
            int kk  = idx & 31;
            As[i][kk] = (_Float16)x[(size_t)(row0 + i) * DM + (k0 + kk)];
        }
        // stage B transposed: W[k0..+32][col0..+64]
        #pragma unroll
        for (int r = 0; r < 8; ++r) {
            int idx = tid + r * 256;            // 0..2047
            int kk  = idx >> 6;
            int nn  = idx & 63;
            Bs[nn][kk] = (_Float16)W[(size_t)(k0 + kk) * DM + (col0 + nn)];
        }
        __syncthreads();

        v16h a[2], b[2];
        #pragma unroll
        for (int mt = 0; mt < 2; ++mt) {
            const _Float16* p = &As[wm * 32 + mt * 16 + lr][kq * 16];
            v8h lo = *(const v8h*)p;
            v8h hi = *(const v8h*)(p + 8);
            a[mt] = __builtin_shufflevector(lo, hi, CAT16);
        }
        #pragma unroll
        for (int nt = 0; nt < 2; ++nt) {
            const _Float16* p = &Bs[wn * 32 + nt * 16 + lr][kq * 16];
            v8h lo = *(const v8h*)p;
            v8h hi = *(const v8h*)(p + 8);
            b[nt] = __builtin_shufflevector(lo, hi, CAT16);
        }
        #pragma unroll
        for (int mt = 0; mt < 2; ++mt)
            #pragma unroll
            for (int nt = 0; nt < 2; ++nt)
                acc[mt][nt] = __builtin_amdgcn_wmma_f32_16x16x32_f16(
                    false, a[mt], false, b[nt], (short)0, acc[mt][nt], false, false);
        __syncthreads();
    }

    // epilogue: D layout -> m = r + 8*(lane/16), n = lane%16
    const int lh = lane >> 4;
    #pragma unroll
    for (int mt = 0; mt < 2; ++mt)
        #pragma unroll
        for (int nt = 0; nt < 2; ++nt)
            #pragma unroll
            for (int r = 0; r < 8; ++r) {
                int row = row0 + wm * 32 + mt * 16 + r + lh * 8;
                int col = col0 + wn * 32 + nt * 16 + lr;
                float val = acc[mt][nt][r] + bias[col];
                size_t o = (size_t)row * DM + col;
                if (mode == 0)      sigq[o] = 1.0f / (1.0f + __expf(-val));
                else if (mode == 1) kbuf[o] = (_Float16)val;
                else                vbuf[o] = (_Float16)val;
            }
}

// ---------------------------------------------------------------------------
// Kernel 2: kmax[b][d] = max_n k[b][n][d]
// ---------------------------------------------------------------------------
__global__ void kmax_kernel(const _Float16* __restrict__ kbuf, float* __restrict__ kmax)
{
    int t = blockIdx.x * blockDim.x + threadIdx.x;   // 0..NB*DM-1
    int b = t >> 9;
    int d = t & (DM - 1);
    const _Float16* p = kbuf + (size_t)b * NSEQ * DM + d;
    float m = -1e30f;
    for (int n = 0; n < NSEQ; ++n)
        m = fmaxf(m, (float)p[(size_t)n * DM]);
    kmax[t] = m;
}

// ---------------------------------------------------------------------------
// Kernel 3: per-row max of pos_bias, eb = exp(pos_bias - rowmax) -> f16
// ---------------------------------------------------------------------------
__global__ void __launch_bounds__(256)
eb_kernel(const float* __restrict__ pos_bias, _Float16* __restrict__ ebuf)
{
    __shared__ float red[256];
    const int i = blockIdx.x;
    const float* row = pos_bias + (size_t)i * NSEQ;
    float m = -1e30f;
    for (int j = threadIdx.x; j < NSEQ; j += 256) m = fmaxf(m, row[j]);
    red[threadIdx.x] = m;
    __syncthreads();
    for (int s = 128; s > 0; s >>= 1) {
        if (threadIdx.x < s)
            red[threadIdx.x] = fmaxf(red[threadIdx.x], red[threadIdx.x + s]);
        __syncthreads();
    }
    const float bm = red[0];
    _Float16* orow = ebuf + (size_t)i * NSEQ;
    for (int j = threadIdx.x; j < NSEQ; j += 256)
        orow[j] = (_Float16)__expf(row[j] - bm);
}

// ---------------------------------------------------------------------------
// Kernel 4: ek = exp(k - kmax); p = ek * v   (in-place over kbuf / vbuf)
// ---------------------------------------------------------------------------
__global__ void prep_kernel(_Float16* kb, _Float16* vb, const float* __restrict__ kmax)
{
    size_t t = (size_t)blockIdx.x * blockDim.x + threadIdx.x;
    int d = (int)(t & (DM - 1));
    int b = (int)(t >> 20);                       // NSEQ*DM = 2^20 per batch
    float km = kmax[(b << 9) + d];
    float e  = __expf((float)kb[t] - km);
    float p  = e * (float)vb[t];
    kb[t] = (_Float16)e;
    vb[t] = (_Float16)p;
}

// ---------------------------------------------------------------------------
// Kernel 5: per batch: num = eb @ p, den = eb @ ek (shared A tile, dual acc);
//           out = sigmoid(q) * num / den
// Double-buffered LDS; A tile staged with gfx1250 async global->LDS copies.
// ---------------------------------------------------------------------------
__global__ void __launch_bounds__(256)
aft_kernel(const _Float16* __restrict__ ebuf,   // [NSEQ][NSEQ]
           const _Float16* __restrict__ pbuf,   // ek*v [NB][NSEQ][DM]
           const _Float16* __restrict__ ekbuf,  // ek   [NB][NSEQ][DM]
           float* __restrict__ out)             // in: sigmoid(q); out: final
{
    __shared__ alignas(16) _Float16 As[2][MT][LDA];
    __shared__ alignas(16) _Float16 Bp[2][NT][LDA];
    __shared__ alignas(16) _Float16 Be[2][NT][LDA];

    const int tid  = threadIdx.x;
    const int lane = tid & 31;
    const int wave = tid >> 5;
    const int wm   = wave & 3;
    const int wn   = wave >> 2;

    const int i0 = blockIdx.x * MT;
    const int d0 = blockIdx.y * NT;
    const int b  = blockIdx.z;
    const _Float16* pb = pbuf  + (size_t)b * NSEQ * DM;
    const _Float16* ek = ekbuf + (size_t)b * NSEQ * DM;

    v8f accP[2][2] = {}, accE[2][2] = {};

    const int kq = lane >> 4;
    const int lr = lane & 15;

    // stage A tile (pure f16 copy): async global->LDS when available
    auto stageA = [&](int buf, int j0) {
        #pragma unroll
        for (int r = 0; r < 2; ++r) {
            int idx = tid + r * 256;             // 0..511 b128 chunks
            int i   = idx >> 2;                  // tile row 0..127
            int c   = (idx & 3) * 8;             // half offset within row
            const _Float16* g = &ebuf[(size_t)(i0 + i) * NSEQ + j0 + c];
#if HAVE_ASYNC_LDS
            __builtin_amdgcn_global_load_async_to_lds_b128(
                (gv4i*)g, (lv4i*)&As[buf][i][c], 0, 0);
#else
            *(v8h*)&As[buf][i][c] = *(const v8h*)g;
#endif
        }
    };
    // stage B tiles transposed (needs VALU shuffle, stays on normal path)
    auto stageB = [&](int buf, int j0) {
        #pragma unroll
        for (int r = 0; r < 4; ++r) {
            int idx = tid + r * 256;             // 0..1023
            int kk  = idx >> 5;                  // j row 0..31
            int n2  = (idx & 31) * 2;            // d col pair
            v2h vp = *(const v2h*)&pb[(size_t)(j0 + kk) * DM + d0 + n2];
            v2h ve = *(const v2h*)&ek[(size_t)(j0 + kk) * DM + d0 + n2];
            Bp[buf][n2][kk] = vp.x; Bp[buf][n2 + 1][kk] = vp.y;
            Be[buf][n2][kk] = ve.x; Be[buf][n2 + 1][kk] = ve.y;
        }
    };

    stageA(0, 0);
    stageB(0, 0);
#if HAVE_ASYNC_LDS
    __builtin_amdgcn_s_wait_asynccnt(0);
#endif
    __syncthreads();

    for (int j0 = 0; j0 < NSEQ; j0 += KSTEP) {
        const int cur = (j0 >> 5) & 1;
        const int nxt = cur ^ 1;

        // prefetch B slab two steps ahead into cache (gfx1250 global_prefetch_b8)
        if (j0 + 2 * KSTEP < NSEQ) {
            __builtin_prefetch(&pb[(size_t)(j0 + 2 * KSTEP + (tid & 31)) * DM + d0], 0, 1);
            __builtin_prefetch(&ek[(size_t)(j0 + 2 * KSTEP + (tid & 31)) * DM + d0], 0, 1);
        }
        // stage next tiles while computing on current ones
        if (j0 + KSTEP < NSEQ) {
            stageA(nxt, j0 + KSTEP);
            stageB(nxt, j0 + KSTEP);
        }

        v16h a[2], bpf[2], bef[2];
        #pragma unroll
        for (int mt = 0; mt < 2; ++mt) {
            const _Float16* p = &As[cur][wm * 32 + mt * 16 + lr][kq * 16];
            v8h lo = *(const v8h*)p;
            v8h hi = *(const v8h*)(p + 8);
            a[mt] = __builtin_shufflevector(lo, hi, CAT16);
        }
        #pragma unroll
        for (int nt = 0; nt < 2; ++nt) {
            const _Float16* pp = &Bp[cur][wn * 32 + nt * 16 + lr][kq * 16];
            v8h plo = *(const v8h*)pp;
            v8h phi = *(const v8h*)(pp + 8);
            bpf[nt] = __builtin_shufflevector(plo, phi, CAT16);
            const _Float16* pe = &Be[cur][wn * 32 + nt * 16 + lr][kq * 16];
            v8h elo = *(const v8h*)pe;
            v8h ehi = *(const v8h*)(pe + 8);
            bef[nt] = __builtin_shufflevector(elo, ehi, CAT16);
        }
        #pragma unroll
        for (int mt = 0; mt < 2; ++mt)
            #pragma unroll
            for (int nt = 0; nt < 2; ++nt) {
                accP[mt][nt] = __builtin_amdgcn_wmma_f32_16x16x32_f16(
                    false, a[mt], false, bpf[nt], (short)0, accP[mt][nt], false, false);
                accE[mt][nt] = __builtin_amdgcn_wmma_f32_16x16x32_f16(
                    false, a[mt], false, bef[nt], (short)0, accE[mt][nt], false, false);
            }

#if HAVE_ASYNC_LDS
        __builtin_amdgcn_s_wait_asynccnt(0);
#endif
        __syncthreads();
    }

    const int lh = lane >> 4;
    #pragma unroll
    for (int mt = 0; mt < 2; ++mt)
        #pragma unroll
        for (int nt = 0; nt < 2; ++nt)
            #pragma unroll
            for (int r = 0; r < 8; ++r) {
                int i = i0 + wm * 32 + mt * 16 + r + lh * 8;
                int d = d0 + wn * 32 + nt * 16 + lr;
                size_t o = ((size_t)b * NSEQ + i) * DM + d;
                float num = accP[mt][nt][r];
                float den = accE[mt][nt][r];
                out[o] = out[o] * (num / den);
            }
}

// ---------------------------------------------------------------------------
extern "C" void kernel_launch(void* const* d_in, const int* in_sizes, int n_in,
                              void* d_out, int out_size, void* d_ws, size_t ws_size,
                              hipStream_t stream)
{
    const float* x        = (const float*)d_in[0];
    const float* Wq       = (const float*)d_in[1];
    const float* bq       = (const float*)d_in[2];
    const float* Wk       = (const float*)d_in[3];
    const float* bk       = (const float*)d_in[4];
    const float* Wv       = (const float*)d_in[5];
    const float* bv       = (const float*)d_in[6];
    const float* pos_bias = (const float*)d_in[7];
    float* out = (float*)d_out;

    char* w = (char*)d_ws;
    const size_t S = (size_t)NB * NSEQ * DM;               // 16,777,216 elems
    _Float16* kbuf = (_Float16*)w; w += S * sizeof(_Float16);              // 32 MB
    _Float16* vbuf = (_Float16*)w; w += S * sizeof(_Float16);              // 32 MB
    _Float16* ebuf = (_Float16*)w; w += (size_t)NSEQ * NSEQ * sizeof(_Float16); // 8 MB
    float*    kmax = (float*)w;                                            // 32 KB

    dim3 blk(256);
    qkv_kernel<<<dim3((NB * NSEQ) / MT, DM / NT, 3), blk, 0, stream>>>(
        x, Wq, bq, Wk, bk, Wv, bv, out, kbuf, vbuf);
    kmax_kernel<<<dim3((NB * DM) / 256), blk, 0, stream>>>(kbuf, kmax);
    eb_kernel<<<dim3(NSEQ), blk, 0, stream>>>(pos_bias, ebuf);
    prep_kernel<<<dim3((unsigned)(S / 256)), blk, 0, stream>>>(kbuf, vbuf, kmax);
    aft_kernel<<<dim3(NSEQ / MT, DM / NT, NB), blk, 0, stream>>>(ebuf, vbuf, kbuf, out);
}
